// Mamba_5042291605980
// MI455X (gfx1250) — matrix operand order repaired and checked by reading, not compiled, not run on hardware
//
#include <hip/hip_runtime.h>

// ---------------- model dims ----------------
#define BH    2
#define LH    1024
#define DMOD  768
#define DI    1536
#define DS    16
#define DTR   48
#define DTRP  64      // dt_rank padded to a WMMA-K multiple of 32
#define NL    4
#define NV    32000
#define DC    4
#define MROWS (BH*LH) // 2048 token rows

// ---------------- CDNA5 WMMA types ----------------
typedef __attribute__((ext_vector_type(16))) __bf16 v16bf;
typedef __attribute__((ext_vector_type(8)))  __bf16 v8bf;
typedef __attribute__((ext_vector_type(8)))  float  v8f;

static __device__ __forceinline__ unsigned short f2bf(float f) {
  unsigned int u = __float_as_uint(f);
  u += 0x7fffu + ((u >> 16) & 1u);          // round-to-nearest-even
  return (unsigned short)(u >> 16);
}

// A tile (16x32 bf16, row-major source): lanes 0-15 hold K 0..15,
// lanes 16-31 hold K 16..31 -> two contiguous 16-byte groups.
static __device__ __forceinline__ v16bf load_a_tile(const unsigned short* p) {
  v8bf g0 = *reinterpret_cast<const v8bf*>(p);
  v8bf g1 = *reinterpret_cast<const v8bf*>(p + 8);
  return __builtin_shufflevector(g0, g1, 0, 1, 2, 3, 4, 5, 6, 7,
                                 8, 9, 10, 11, 12, 13, 14, 15);
}
// B tile (32x16 bf16): lane holds 16 contiguous K of one column (= one row of W[N][K]).
static __device__ __forceinline__ v16bf load_b_tile(const unsigned short* p) {
  return *reinterpret_cast<const v16bf*>(p);
}
static __device__ __forceinline__ v8f wmma_bf16(v16bf a, v16bf b, v8f c) {
  return __builtin_amdgcn_wmma_f32_16x16x32_bf16(false, a, false, b, (short)0, c,
                                                 false, false);
}

// ---------------- GEMM: C[M,N] = A[M,K](bf16) * W[N,K]^T(bf16) (+bias)(+resid)(softplus) ----
// One wave per block; wave computes a 16 x (NT*16) strip starting at column
// nbase + blockIdx.y*NT*16. Branch-free K-loop (NT is a compile-time constant)
// so WMMA accumulators stay register-resident (no spills, EXEC always full).
template <int NT>
__global__ void gemm_bf16_wmma_kernel(const unsigned short* __restrict__ A, int lda,
                                      const unsigned short* __restrict__ W, int ldb,
                                      float* C, int ldc, int K, int nbase,
                                      const float* __restrict__ bias,
                                      const float* resid, int act) {
  const int lane = threadIdx.x & 31;
  const int m0 = blockIdx.x << 4;
  const int n0 = nbase + blockIdx.y * (NT * 16);
  const int mi = lane & 15;
  const int hi = lane >> 4;

  v8f acc[NT];
#pragma unroll
  for (int t = 0; t < NT; ++t)
    acc[t] = (v8f){0.f, 0.f, 0.f, 0.f, 0.f, 0.f, 0.f, 0.f};

  const unsigned short* arow = A + (size_t)(m0 + mi) * lda + hi * 16;
  const unsigned short* brow = W + (size_t)(n0 + mi) * ldb + hi * 16;
  for (int k = 0; k < K; k += 32) {
    v16bf a = load_a_tile(arow + k);
#pragma unroll
    for (int t = 0; t < NT; ++t) {
      v16bf b = load_b_tile(brow + (size_t)(t * 16) * ldb + k);
      acc[t] = wmma_bf16(a, b, acc[t]);
    }
  }
  // C/D layout: VGPR r -> M = m0 + 8*hi + r ; N = n0 + 16t + (lane&15)
  const int mr = m0 + hi * 8;
#pragma unroll
  for (int t = 0; t < NT; ++t) {
    const int col = n0 + t * 16 + mi;
    const float bterm = bias ? bias[col] : 0.f;
#pragma unroll
    for (int r = 0; r < 8; ++r) {
      size_t idx = (size_t)(mr + r) * ldc + col;
      float v = acc[t][r] + bterm;
      if (resid) v += resid[idx];
      if (act == 1) v = (v > 20.f) ? v : log1pf(__expf(v));  // softplus
      C[idx] = v;
    }
  }
}

// ---------------- helpers ----------------
__global__ void convert_bf16_kernel(const float* __restrict__ s, unsigned short* d, int n) {
  int i = blockIdx.x * blockDim.x + threadIdx.x;
  if (i < n) d[i] = f2bf(s[i]);
}
__global__ void pad_convert_bf16_kernel(const float* __restrict__ s, int sstride, int scols,
                                        unsigned short* d, int dcols, int n) {
  int i = blockIdx.x * blockDim.x + threadIdx.x;
  if (i >= n) return;
  int r = i / dcols, c = i - r * dcols;
  d[i] = (c < scols) ? f2bf(s[(size_t)r * sstride + c]) : (unsigned short)0;
}
__global__ void embed_kernel(const int* __restrict__ ids, const float* __restrict__ emb,
                             float* x, int n) {
  int i = blockIdx.x * blockDim.x + threadIdx.x;
  if (i >= n) return;
  int row = i / DMOD, c = i - row * DMOD;
  x[i] = emb[(size_t)ids[row] * DMOD + c];
}
__global__ void rmsnorm_bf16_kernel(const float* __restrict__ x, const float* __restrict__ w,
                                    unsigned short* out, int ncols) {
  __shared__ float sred[256];
  const int row = blockIdx.x;
  const float* xr = x + (size_t)row * ncols;
  float s = 0.f;
  for (int c = threadIdx.x; c < ncols; c += 256) { float v = xr[c]; s += v * v; }
  sred[threadIdx.x] = s;
  __syncthreads();
  for (int st = 128; st > 0; st >>= 1) {
    if ((int)threadIdx.x < st) sred[threadIdx.x] += sred[threadIdx.x + st];
    __syncthreads();
  }
  const float scale = rsqrtf(sred[0] / (float)ncols + 1e-5f);
  for (int c = threadIdx.x; c < ncols; c += 256)
    out[(size_t)row * ncols + c] = f2bf(xr[c] * scale * w[c]);
}
// causal depthwise conv (width 4) + SiLU; input = xin half of xz
__global__ void conv_silu_kernel(const float* __restrict__ xz, const float* __restrict__ cw,
                                 const float* __restrict__ cb, float* u,
                                 unsigned short* ubf, int n) {
  int i = blockIdx.x * blockDim.x + threadIdx.x;
  if (i >= n) return;
  int d = i % DI, row = i / DI, l = row % LH;
  const float* w = cw + d * DC;
  float s = cb[d];
#pragma unroll
  for (int j = 0; j < DC; ++j) {
    int ll = l - (DC - 1) + j;
    if (ll >= 0) s += xz[(size_t)(row - l + ll) * (2 * DI) + d] * w[j];
  }
  s = s / (1.f + __expf(-s));
  u[i] = s;
  ubf[i] = f2bf(s);
}
// selective scan: one thread per (b,d); 16 states in registers, sequential over L
__global__ void scan_kernel(const float* __restrict__ u, const float* __restrict__ delta,
                            const float* __restrict__ xdbl, const float* __restrict__ A_log,
                            const float* __restrict__ Dp, float* y) {
  int t = blockIdx.x * blockDim.x + threadIdx.x;
  if (t >= BH * DI) return;
  int b = t / DI, d = t - b * DI;
  float A[DS], h[DS];
#pragma unroll
  for (int n = 0; n < DS; ++n) { A[n] = -__expf(A_log[d * DS + n]); h[n] = 0.f; }
  const float Dd = Dp[d];
  for (int l = 0; l < LH; ++l) {
    size_t row = (size_t)b * LH + l;
    float ut = u[row * DI + d];
    float dt = delta[row * DI + d];
    const float* xr = xdbl + row * (DTR + 2 * DS);
    float acc = 0.f;
#pragma unroll
    for (int n = 0; n < DS; ++n) {
      h[n] = __expf(dt * A[n]) * h[n] + dt * xr[DTR + n] * ut;
      acc += h[n] * xr[DTR + DS + n];
    }
    y[row * DI + d] = acc + ut * Dd;
  }
}
__global__ void gate_kernel(const float* __restrict__ y, const float* __restrict__ xz,
                            unsigned short* ybf, int n) {
  int i = blockIdx.x * blockDim.x + threadIdx.x;
  if (i >= n) return;
  int d = i % DI, row = i / DI;
  float r = xz[(size_t)row * (2 * DI) + DI + d];
  float g = r / (1.f + __expf(-r));
  ybf[i] = f2bf(y[i] * g);
}

// ---------------- driver ----------------
extern "C" void kernel_launch(void* const* d_in, const int* in_sizes, int n_in,
                              void* d_out, int out_size, void* d_ws, size_t ws_size,
                              hipStream_t stream) {
  const int*   ids   = (const int*)d_in[0];
  const float* emb   = (const float*)d_in[4];
  const float* normw = (const float*)d_in[5];
  const float* inw   = (const float*)d_in[6];
  const float* convw = (const float*)d_in[7];
  const float* convb = (const float*)d_in[8];
  const float* xpw   = (const float*)d_in[9];
  const float* dtw   = (const float*)d_in[10];
  const float* dtb   = (const float*)d_in[11];
  const float* alog  = (const float*)d_in[12];
  const float* dpar  = (const float*)d_in[13];
  const float* outw  = (const float*)d_in[14];
  const float* normf = (const float*)d_in[15];
  float* logits = (float*)d_out;
  (void)in_sizes; (void)n_in; (void)out_size; (void)ws_size;

  char* ws = (char*)d_ws;
  size_t off = 0;
  auto alloc = [&](size_t bytes) -> char* {
    char* p = ws + off;
    off = (off + bytes + 255) & ~(size_t)255;
    return p;
  };
  unsigned short* emb_bf  = (unsigned short*)alloc((size_t)NV * DMOD * 2);
  unsigned short* inw_bf  = (unsigned short*)alloc((size_t)NL * 2 * DI * DMOD * 2);
  unsigned short* xpw_bf  = (unsigned short*)alloc((size_t)NL * (DTR + 2 * DS) * DI * 2);
  unsigned short* dtw_bf  = (unsigned short*)alloc((size_t)NL * DI * DTRP * 2);
  unsigned short* outw_bf = (unsigned short*)alloc((size_t)NL * DMOD * DI * 2);
  float*          x       = (float*)alloc((size_t)MROWS * DMOD * 4);
  unsigned short* xn_bf   = (unsigned short*)alloc((size_t)MROWS * DMOD * 2);
  float*          xz      = (float*)alloc((size_t)MROWS * 2 * DI * 4);
  float*          u       = (float*)alloc((size_t)MROWS * DI * 4);
  unsigned short* u_bf    = (unsigned short*)alloc((size_t)MROWS * DI * 2);
  float*          xdbl    = (float*)alloc((size_t)MROWS * (DTR + 2 * DS) * 4);
  unsigned short* dlt_bf  = (unsigned short*)alloc((size_t)MROWS * DTRP * 2);
  float*          delta   = (float*)alloc((size_t)MROWS * DI * 4);
  float*          yb      = (float*)alloc((size_t)MROWS * DI * 4);
  unsigned short* y_bf    = (unsigned short*)alloc((size_t)MROWS * DI * 2);

  auto cdiv = [](int a, int b) { return (a + b - 1) / b; };

  // GEMM dispatcher: 64-wide strips (branch-free NT=4), 16-wide tail for N%64.
  auto launch_gemm = [&](const unsigned short* A, int lda, const unsigned short* W, int ldb,
                         float* C, int ldc, int N, int K,
                         const float* bias, const float* resid, int act) {
    int n64 = (N / 64) * 64;
    if (n64 > 0)
      gemm_bf16_wmma_kernel<4><<<dim3(MROWS / 16, n64 / 64), 32, 0, stream>>>(
          A, lda, W, ldb, C, ldc, K, 0, bias, resid, act);
    for (int n = n64; n < N; n += 16)
      gemm_bf16_wmma_kernel<1><<<dim3(MROWS / 16, 1), 32, 0, stream>>>(
          A, lda, W, ldb, C, ldc, K, n, bias, resid, act);
  };

  // weights -> bf16 (every call; deterministic; bf16 copies are L2-resident)
  { int n = NV * DMOD;            convert_bf16_kernel<<<cdiv(n,256),256,0,stream>>>(emb,  emb_bf,  n); }
  { int n = NL * 2 * DI * DMOD;   convert_bf16_kernel<<<cdiv(n,256),256,0,stream>>>(inw,  inw_bf,  n); }
  { int n = NL * (DTR+2*DS) * DI; convert_bf16_kernel<<<cdiv(n,256),256,0,stream>>>(xpw,  xpw_bf,  n); }
  { int n = NL * DI * DTRP;       pad_convert_bf16_kernel<<<cdiv(n,256),256,0,stream>>>(dtw, DTR, DTR, dtw_bf, DTRP, n); }
  { int n = NL * DMOD * DI;       convert_bf16_kernel<<<cdiv(n,256),256,0,stream>>>(outw, outw_bf, n); }

  { int n = MROWS * DMOD; embed_kernel<<<cdiv(n,256),256,0,stream>>>(ids, emb, x, n); }

  for (int layer = 0; layer < NL; ++layer) {
    rmsnorm_bf16_kernel<<<MROWS, 256, 0, stream>>>(x, normw + layer * DMOD, xn_bf, DMOD);
    // in_proj: (2048x768) x (3072x768)^T -> xz
    launch_gemm(xn_bf, DMOD, inw_bf + (size_t)layer * 2 * DI * DMOD, DMOD,
                xz, 2 * DI, 2 * DI, DMOD, nullptr, nullptr, 0);
    { int n = MROWS * DI;
      conv_silu_kernel<<<cdiv(n,256),256,0,stream>>>(xz, convw + layer*DI*DC, convb + layer*DI, u, u_bf, n); }
    // x_proj: (2048x1536) x (80x1536)^T -> xdbl  (one 64-strip + one 16-tail)
    launch_gemm(u_bf, DI, xpw_bf + (size_t)layer * (DTR+2*DS) * DI, DI,
                xdbl, DTR + 2*DS, DTR + 2*DS, DI, nullptr, nullptr, 0);
    { int n = MROWS * DTRP;
      pad_convert_bf16_kernel<<<cdiv(n,256),256,0,stream>>>(xdbl, DTR+2*DS, DTR, dlt_bf, DTRP, n); }
    // dt_proj + bias + softplus: (2048x64) x (1536x64)^T -> delta
    launch_gemm(dlt_bf, DTRP, dtw_bf + (size_t)layer * DI * DTRP, DTRP,
                delta, DI, DI, DTRP, dtb + layer * DI, nullptr, 1);
    scan_kernel<<<cdiv(BH*DI,256),256,0,stream>>>(u, delta, xdbl,
        alog + (size_t)layer * DI * DS, dpar + layer * DI, yb);
    { int n = MROWS * DI; gate_kernel<<<cdiv(n,256),256,0,stream>>>(yb, xz, y_bf, n); }
    // out_proj with fused residual: x = y @ W^T + x
    launch_gemm(y_bf, DI, outw_bf + (size_t)layer * DMOD * DI, DI,
                x, DMOD, DMOD, DI, nullptr, x, 0);
  }
  rmsnorm_bf16_kernel<<<MROWS, 256, 0, stream>>>(x, normf, xn_bf, DMOD);
  // tied lm_head: (2048x768) x (32000x768)^T -> logits
  launch_gemm(xn_bf, DMOD, emb_bf, DMOD, logits, NV, NV, DMOD, nullptr, nullptr, 0);
}
